// MyTransformer_53927609368766
// MI455X (gfx1250) — compile-verified, matrix operand order checked by reading, not run on hardware
//
#include <hip/hip_runtime.h>
#include <hip/hip_bf16.h>

typedef __attribute__((ext_vector_type(16))) __bf16 v16bf;
typedef __attribute__((ext_vector_type(8)))  __bf16 v8bf;
typedef __attribute__((ext_vector_type(4)))  __bf16 v4bf;
typedef __attribute__((ext_vector_type(8)))  float  v8f;

#define D_MODEL 1024
#define T_SEQ   1024
#define N_HEAD  16
#define HEAD_D  64
#define D_KV    256
#define N_LAYER 12
#define D_FF    4096
#define N_VOCAB 32000
#define RANK    16
#define MT      2048          // B*T tokens
#define LORA_SCALE 2.0f

// K-index map for 16-bit A/B WMMA fragments (ISA 7.12.2):
// lanes 0-15 hold K {0..7,16..23}, lanes 16-31 hold K {8..15,24..31}.
// Per lane that is two contiguous 8-element groups at k = hi*8 and hi*8+16.

__device__ __forceinline__ v8bf pack8(const float* f) {
  v8bf r;
#pragma unroll
  for (int i = 0; i < 8; ++i) r[i] = (__bf16)f[i];
  return r;
}

__device__ __forceinline__ v16bf pack16(float4 a, float4 b, float4 c, float4 d) {
  v16bf r;
  r[0]=(__bf16)a.x; r[1]=(__bf16)a.y; r[2]=(__bf16)a.z; r[3]=(__bf16)a.w;
  r[4]=(__bf16)b.x; r[5]=(__bf16)b.y; r[6]=(__bf16)b.z; r[7]=(__bf16)b.w;
  r[8]=(__bf16)c.x; r[9]=(__bf16)c.y; r[10]=(__bf16)c.z; r[11]=(__bf16)c.w;
  r[12]=(__bf16)d.x; r[13]=(__bf16)d.y; r[14]=(__bf16)d.z; r[15]=(__bf16)d.w;
  return r;
}

__device__ __forceinline__ v16bf join16(v8bf lo, v8bf hi) {
  return __builtin_shufflevector(lo, hi, 0,1,2,3,4,5,6,7,8,9,10,11,12,13,14,15);
}

// ---------------------------------------------------------------- embedding
__global__ void embed_kernel(const int* __restrict__ idx,
                             const float* __restrict__ tok,
                             const float* __restrict__ pos,
                             float* __restrict__ x) {
  int i = blockIdx.x * blockDim.x + threadIdx.x;   // over MT*D
  int row = i >> 10;
  int c = i & (D_MODEL - 1);
  int t = row & (T_SEQ - 1);
  int tid = idx[row];
  x[i] = tok[(size_t)tid * D_MODEL + c] + pos[(size_t)t * D_MODEL + c];
}

// ---------------------------------------------------------------- layernorm
__global__ __launch_bounds__(256) void ln_kernel(const float* __restrict__ x,
                                                 const float* __restrict__ w,
                                                 const float* __restrict__ b,
                                                 float* __restrict__ y) {
  __shared__ float red[256];
  const int row = blockIdx.x, tid = threadIdx.x;
  const float* xr = x + (size_t)row * D_MODEL;
  float loc[4], s = 0.f;
#pragma unroll
  for (int i = 0; i < 4; ++i) { loc[i] = xr[tid + i * 256]; s += loc[i]; }
  red[tid] = s; __syncthreads();
  for (int st = 128; st > 0; st >>= 1) { if (tid < st) red[tid] += red[tid + st]; __syncthreads(); }
  float mean = red[0] * (1.0f / D_MODEL);
  __syncthreads();
  float vs = 0.f;
#pragma unroll
  for (int i = 0; i < 4; ++i) { float d = loc[i] - mean; vs += d * d; }
  red[tid] = vs; __syncthreads();
  for (int st = 128; st > 0; st >>= 1) { if (tid < st) red[tid] += red[tid + st]; __syncthreads(); }
  float inv = rsqrtf(red[0] * (1.0f / D_MODEL) + 1e-5f);
#pragma unroll
  for (int i = 0; i < 4; ++i) {
    int c = tid + i * 256;
    y[(size_t)row * D_MODEL + c] = (loc[i] - mean) * inv * w[c] + b[c];
  }
}

// ---------------------------------------------------------------- LoRA
__global__ void lora_xa_kernel(const float* __restrict__ h,
                               const float* __restrict__ A,
                               float* __restrict__ t) {
  int i = blockIdx.x * blockDim.x + threadIdx.x;   // over MT*RANK
  int m = i >> 4, r = i & 15;
  const float* hr = h + (size_t)m * D_MODEL;
  float s = 0.f;
  for (int k = 0; k < D_MODEL; ++k) s += hr[k] * A[(size_t)k * RANK + r];
  t[i] = s;
}

__global__ void lora_add_kernel(float* __restrict__ out,
                                const float* __restrict__ t,
                                const float* __restrict__ Bw, int N) {
  int i = blockIdx.x * blockDim.x + threadIdx.x;   // over MT*N
  int m = i / N, n = i - m * N;
  float s = 0.f;
#pragma unroll
  for (int r = 0; r < RANK; ++r) s += t[m * RANK + r] * Bw[(size_t)r * N + n];
  out[i] += LORA_SCALE * s;
}

// ---------------------------------------------------------------- V transpose
// v[b][t][d] -> vt[b][d][t]  so attention P@V B-fragments are contiguous.
__global__ void vt_kernel(const float* __restrict__ v, float* __restrict__ vt) {
  int i = blockIdx.x * blockDim.x + threadIdx.x;   // over B*T*DKV
  int d = i & (D_KV - 1);
  int t = (i >> 8) & (T_SEQ - 1);
  int b = i >> 18;
  vt[((size_t)b * D_KV + d) * T_SEQ + t] = v[i];
}

// ---------------------------------------------------------------- WMMA GEMM
// C[M,N] = A[M,K] @ W[K,N] (+bias) (+res) (relu?) ; fp32 in/out, bf16 mma.
// Block: 256 thr = 8 waves (2x4). Tile 128x128, BK=32.
// sA[row][k] and sBt[col][k] both stride-40 (80B rows: 16B-aligned groups,
// 20-bank rotation -> conflict-free fragment gathers as 2x ds_load_b128).
// W tile is transposed in registers: 4x coalesced float4 loads per thread
// (4 k-rows x 4 cols) -> 4x v4bf ds_store_b64 columns. No scalar global
// traffic anywhere in the GEMM.
__global__ __launch_bounds__(256) void gemm_kernel(
    const float* __restrict__ A, const float* __restrict__ W,
    const float* __restrict__ bias, const float* __restrict__ res,
    float* __restrict__ C, int M, int N, int K, int relu) {
  __shared__ alignas(16) __bf16 sA[128][40];
  __shared__ alignas(16) __bf16 sBt[128][40];
  const int tid = threadIdx.x;
  const int lane = tid & 31;
  const int wv = tid >> 5;
  const int wm = wv >> 2, wn = wv & 3;
  const int ln = lane & 15, hi = lane >> 4;
  const int m0 = blockIdx.y * 128, n0 = blockIdx.x * 128;

  // staging assignments
  const int arow = tid >> 1, ac0 = (tid & 1) * 16;       // A: row, 16-col half
  const int br0 = (tid >> 5) * 4;                        // W: 4 k-rows
  const int bc0 = (tid & 31) * 4;                        // W: 4 cols

  v8f acc[4][2] = {};

  for (int k0 = 0; k0 < K; k0 += 32) {
    {  // A tile: 4x float4 loads -> 2x b128 LDS stores
      const float* ap = &A[(size_t)(m0 + arow) * K + k0 + ac0];
      float av[16];
#pragma unroll
      for (int j = 0; j < 4; ++j) {
        float4 f = *(const float4*)(ap + 4 * j);
        av[4*j] = f.x; av[4*j+1] = f.y; av[4*j+2] = f.z; av[4*j+3] = f.w;
      }
      *(v8bf*)&sA[arow][ac0]     = pack8(av);
      *(v8bf*)&sA[arow][ac0 + 8] = pack8(av + 8);
      if (k0 + 32 < K) __builtin_prefetch(ap + 32, 0, 1);
    }
    {  // W tile: 4x coalesced float4 (4 k-rows x 4 cols), transpose in regs
      const float* wp = &W[(size_t)(k0 + br0) * N + n0 + bc0];
      float4 wr[4];
#pragma unroll
      for (int rr = 0; rr < 4; ++rr)
        wr[rr] = *(const float4*)(wp + (size_t)rr * N);
      v4bf c0 = { (__bf16)wr[0].x, (__bf16)wr[1].x, (__bf16)wr[2].x, (__bf16)wr[3].x };
      v4bf c1 = { (__bf16)wr[0].y, (__bf16)wr[1].y, (__bf16)wr[2].y, (__bf16)wr[3].y };
      v4bf c2 = { (__bf16)wr[0].z, (__bf16)wr[1].z, (__bf16)wr[2].z, (__bf16)wr[3].z };
      v4bf c3 = { (__bf16)wr[0].w, (__bf16)wr[1].w, (__bf16)wr[2].w, (__bf16)wr[3].w };
      *(v4bf*)&sBt[bc0 + 0][br0] = c0;
      *(v4bf*)&sBt[bc0 + 1][br0] = c1;
      *(v4bf*)&sBt[bc0 + 2][br0] = c2;
      *(v4bf*)&sBt[bc0 + 3][br0] = c3;
      if (k0 + 32 < K) __builtin_prefetch(wp + (size_t)32 * N, 0, 1);
    }
    __syncthreads();

    v16bf af[4], bfr[2];
#pragma unroll
    for (int fm = 0; fm < 4; ++fm) {
      const __bf16* p = &sA[wm * 64 + fm * 16 + ln][hi * 8];
      af[fm] = join16(*(const v8bf*)p, *(const v8bf*)(p + 16));
    }
#pragma unroll
    for (int fn = 0; fn < 2; ++fn) {
      const __bf16* p = &sBt[wn * 32 + fn * 16 + ln][hi * 8];
      bfr[fn] = join16(*(const v8bf*)p, *(const v8bf*)(p + 16));
    }
#pragma unroll
    for (int fm = 0; fm < 4; ++fm)
#pragma unroll
      for (int fn = 0; fn < 2; ++fn)
        acc[fm][fn] = __builtin_amdgcn_wmma_f32_16x16x32_bf16(
            false, af[fm], false, bfr[fn], (short)0, acc[fm][fn], false, false);
    __syncthreads();
  }

#pragma unroll
  for (int fm = 0; fm < 4; ++fm)
#pragma unroll
    for (int fn = 0; fn < 2; ++fn)
#pragma unroll
      for (int r = 0; r < 8; ++r) {
        int m = m0 + wm * 64 + fm * 16 + (hi << 3) + r;
        int n = n0 + wn * 32 + fn * 16 + ln;
        if (m < M && n < N) {
          float v = acc[fm][fn][r];
          if (bias) v += bias[n];
          if (res)  v += res[(size_t)m * N + n];
          if (relu) v = fmaxf(v, 0.0f);
          C[(size_t)m * N + n] = v;
        }
      }
}

// ---------------------------------------------------------------- attention
// Flash-style causal GQA. Grid: (B*N_HEAD, T/64). Block 128 thr = 4 waves,
// each wave owns 16 query rows. Per 32-key tile: 4 wmma (QK^T) + 4 wmma (PV).
// All global fragment gathers are contiguous float4 loads (V pre-transposed).
__global__ __launch_bounds__(128) void attn_kernel(
    const float* __restrict__ Q, const float* __restrict__ Kb,
    const float* __restrict__ Vt, float* __restrict__ O) {
  const int bh = blockIdx.x;
  const int b = bh >> 4;            // / N_HEAD
  const int h = bh & 15;
  const int g = h >> 2;             // QPG = 4
  const int qt = blockIdx.y;
  const int tid = threadIdx.x;
  const int wv = tid >> 5;
  const int lane = tid & 31;
  const int ln = lane & 15, hi = lane >> 4;
  const int qbase = qt * 64 + wv * 16;
  const size_t bT = (size_t)b * T_SEQ;
  const int hof = h * HEAD_D, gof = g * HEAD_D;

  __shared__ alignas(16) __bf16 sP[4][16][40];  // per-wave P re-layout buffer

  // Q A-fragments (16 rows x 64 head-dim -> two 16x32 frags), contiguous loads
  const float* qp = &Q[(bT + qbase + ln) * D_MODEL + hof];
  v16bf qf[2];
#pragma unroll
  for (int kk = 0; kk < 2; ++kk) {
    const float* p = qp + kk * 32 + hi * 8;
    qf[kk] = pack16(*(const float4*)(p),      *(const float4*)(p + 4),
                    *(const float4*)(p + 16), *(const float4*)(p + 20));
  }

  v8f oacc[4] = {};
  float mrun[8], lrun[8];
#pragma unroll
  for (int r = 0; r < 8; ++r) { mrun[r] = -1e30f; lrun[r] = 0.f; }

  const int ntiles = (qbase + 47) >> 5;   // keys up to qbase+15, 32 per tile
  for (int t = 0; t < ntiles; ++t) {
    const int kb = t * 32;
    // S = Q @ K^T : lane = key column, elements over head-dim (contiguous)
    v8f sc[2] = {};
#pragma unroll
    for (int fn = 0; fn < 2; ++fn) {
      const float* kp = &Kb[(bT + kb + fn * 16 + ln) * D_KV + gof];
#pragma unroll
      for (int kk = 0; kk < 2; ++kk) {
        const float* p = kp + kk * 32 + hi * 8;
        v16bf bk = pack16(*(const float4*)(p),      *(const float4*)(p + 4),
                          *(const float4*)(p + 16), *(const float4*)(p + 20));
        sc[fn] = __builtin_amdgcn_wmma_f32_16x16x32_bf16(
            false, qf[kk], false, bk, (short)0, sc[fn], false, false);
      }
    }
    // causal mask + online softmax (row stats via half-wave shfl reductions)
    float pm[2][8];
#pragma unroll
    for (int r = 0; r < 8; ++r) {
      int row = qbase + (hi << 3) + r;
      float s0 = sc[0][r] * 0.125f;             // HD^-0.5 = 1/8
      float s1 = sc[1][r] * 0.125f;
      if (kb + ln > row)      s0 = -1e30f;
      if (kb + 16 + ln > row) s1 = -1e30f;
      float sm = fmaxf(s0, s1);
#pragma unroll
      for (int off = 1; off < 16; off <<= 1)
        sm = fmaxf(sm, __shfl_xor(sm, off, 32));
      float mnew = fmaxf(mrun[r], sm);
      float p0 = __expf(s0 - mnew);
      float p1 = __expf(s1 - mnew);
      float ps = p0 + p1;
#pragma unroll
      for (int off = 1; off < 16; off <<= 1)
        ps += __shfl_xor(ps, off, 32);
      float alpha = __expf(mrun[r] - mnew);
      lrun[r] = lrun[r] * alpha + ps;
      mrun[r] = mnew;
#pragma unroll
      for (int fv = 0; fv < 4; ++fv) oacc[fv][r] *= alpha;
      pm[0][r] = p0; pm[1][r] = p1;
    }
    // C-layout P -> LDS -> A-layout fragment (per-wave region; DS in-order)
#pragma unroll
    for (int fn = 0; fn < 2; ++fn)
#pragma unroll
      for (int r = 0; r < 8; ++r)
        sP[wv][(hi << 3) + r][fn * 16 + ln] = (__bf16)pm[fn][r];
    __threadfence_block();
    const __bf16* pp = &sP[wv][ln][hi * 8];
    v16bf pf = join16(*(const v8bf*)pp, *(const v8bf*)(pp + 16));
    // O += P @ V^T-layout : lane = hd column, elements over key (contiguous)
#pragma unroll
    for (int fv = 0; fv < 4; ++fv) {
      const float* vp = &Vt[((size_t)b * D_KV + gof + fv * 16 + ln) * T_SEQ + kb + hi * 8];
      v16bf bv = pack16(*(const float4*)(vp),      *(const float4*)(vp + 4),
                        *(const float4*)(vp + 16), *(const float4*)(vp + 20));
      oacc[fv] = __builtin_amdgcn_wmma_f32_16x16x32_bf16(
          false, pf, false, bv, (short)0, oacc[fv], false, false);
    }
    __threadfence_block();
  }

#pragma unroll
  for (int fv = 0; fv < 4; ++fv)
#pragma unroll
    for (int r = 0; r < 8; ++r) {
      int row = qbase + (hi << 3) + r;
      O[(bT + row) * D_MODEL + hof + fv * 16 + ln] = oacc[fv][r] / lrun[r];
    }
}

// ---------------------------------------------------------------- driver
extern "C" void kernel_launch(void* const* d_in, const int* in_sizes, int n_in,
                              void* d_out, int out_size, void* d_ws, size_t ws_size,
                              hipStream_t stream) {
  const int*   idx    = (const int*)  d_in[0];
  const float* tok    = (const float*)d_in[1];
  const float* pos    = (const float*)d_in[2];
  const float* ln1_w  = (const float*)d_in[3];
  const float* ln1_b  = (const float*)d_in[4];
  const float* wq_w   = (const float*)d_in[5];
  const float* wq_b   = (const float*)d_in[6];
  const float* wk_w   = (const float*)d_in[7];
  const float* wk_b   = (const float*)d_in[8];
  const float* wv_w   = (const float*)d_in[9];
  const float* wv_b   = (const float*)d_in[10];
  const float* wo_w   = (const float*)d_in[11];
  const float* wo_b   = (const float*)d_in[12];
  const float* lqa    = (const float*)d_in[13];
  const float* lqb    = (const float*)d_in[14];
  const float* lva    = (const float*)d_in[15];
  const float* lvb    = (const float*)d_in[16];
  const float* ln2_w  = (const float*)d_in[17];
  const float* ln2_b  = (const float*)d_in[18];
  const float* f1_w   = (const float*)d_in[19];
  const float* f1_b   = (const float*)d_in[20];
  const float* f2_w   = (const float*)d_in[21];
  const float* f2_b   = (const float*)d_in[22];
  const float* lnf_w  = (const float*)d_in[23];
  const float* lnf_b  = (const float*)d_in[24];
  const float* head_w = (const float*)d_in[25];
  const float* head_b = (const float*)d_in[26];

  float* ws   = (float*)d_ws;
  float* x    = ws;                                  // MT*D
  float* h    = x    + (size_t)MT * D_MODEL;         // MT*D
  float* q    = h    + (size_t)MT * D_MODEL;         // MT*D
  float* att  = q    + (size_t)MT * D_MODEL;         // MT*D
  float* ff   = att  + (size_t)MT * D_MODEL;         // MT*DFF
  float* kbuf = ff   + (size_t)MT * D_FF;            // MT*DKV
  float* vbuf = kbuf + (size_t)MT * D_KV;            // MT*DKV
  float* vtb  = vbuf + (size_t)MT * D_KV;            // MT*DKV (V transposed)
  float* tb   = vtb  + (size_t)MT * D_KV;            // MT*RANK

  embed_kernel<<<(MT * D_MODEL) / 256, 256, 0, stream>>>(idx, tok, pos, x);

  for (int l = 0; l < N_LAYER; ++l) {
    ln_kernel<<<MT, 256, 0, stream>>>(x, ln1_w + (size_t)l * D_MODEL,
                                      ln1_b + (size_t)l * D_MODEL, h);
    gemm_kernel<<<dim3(D_MODEL / 128, MT / 128), 256, 0, stream>>>(
        h, wq_w + (size_t)l * D_MODEL * D_MODEL, wq_b + (size_t)l * D_MODEL,
        nullptr, q, MT, D_MODEL, D_MODEL, 0);
    lora_xa_kernel<<<(MT * RANK) / 256, 256, 0, stream>>>(
        h, lqa + (size_t)l * D_MODEL * RANK, tb);
    lora_add_kernel<<<(MT * D_MODEL) / 256, 256, 0, stream>>>(
        q, tb, lqb + (size_t)l * RANK * D_MODEL, D_MODEL);
    gemm_kernel<<<dim3(D_KV / 128, MT / 128), 256, 0, stream>>>(
        h, wk_w + (size_t)l * D_MODEL * D_KV, wk_b + (size_t)l * D_KV,
        nullptr, kbuf, MT, D_KV, D_MODEL, 0);
    gemm_kernel<<<dim3(D_KV / 128, MT / 128), 256, 0, stream>>>(
        h, wv_w + (size_t)l * D_MODEL * D_KV, wv_b + (size_t)l * D_KV,
        nullptr, vbuf, MT, D_KV, D_MODEL, 0);
    lora_xa_kernel<<<(MT * RANK) / 256, 256, 0, stream>>>(
        h, lva + (size_t)l * D_MODEL * RANK, tb);
    lora_add_kernel<<<(MT * D_KV) / 256, 256, 0, stream>>>(
        vbuf, tb, lvb + (size_t)l * RANK * D_KV, D_KV);
    vt_kernel<<<(MT * D_KV) / 256, 256, 0, stream>>>(vbuf, vtb);
    attn_kernel<<<dim3(2 * N_HEAD, T_SEQ / 64), 128, 0, stream>>>(
        q, kbuf, vtb, att);
    gemm_kernel<<<dim3(D_MODEL / 128, MT / 128), 256, 0, stream>>>(
        att, wo_w + (size_t)l * D_MODEL * D_MODEL, wo_b + (size_t)l * D_MODEL,
        x, x, MT, D_MODEL, D_MODEL, 0);
    ln_kernel<<<MT, 256, 0, stream>>>(x, ln2_w + (size_t)l * D_MODEL,
                                      ln2_b + (size_t)l * D_MODEL, h);
    gemm_kernel<<<dim3(D_FF / 128, MT / 128), 256, 0, stream>>>(
        h, f1_w + (size_t)l * D_MODEL * D_FF, f1_b + (size_t)l * D_FF,
        nullptr, ff, MT, D_FF, D_MODEL, 1);
    gemm_kernel<<<dim3(D_MODEL / 128, MT / 128), 256, 0, stream>>>(
        ff, f2_w + (size_t)l * D_FF * D_MODEL, f2_b + (size_t)l * D_MODEL,
        x, x, MT, D_MODEL, D_FF, 0);
  }

  ln_kernel<<<MT, 256, 0, stream>>>(x, lnf_w, lnf_b, h);
  gemm_kernel<<<dim3(N_VOCAB / 128, MT / 128), 256, 0, stream>>>(
      h, head_w, head_b, nullptr, (float*)d_out, MT, N_VOCAB, D_MODEL, 0);
}
